// DWI_31215822307839
// MI455X (gfx1250) — matrix operand order repaired and checked by reading, not compiled, not run on hardware
//
#include <hip/hip_runtime.h>
#include <math.h>

// Problem constants (from reference): B=4, C1=256, C2=64, N=64*64=4096
#define NB 4
#define C1 256
#define C2 64
#define NS 4096
#define HID 24
#define EPS 1e-5f
// 0.125 = 1/sqrt(C2) softmax scale, folded with log2(e) so energies are
// base-2 logits and softmax exp is a bare v_exp_f32 (natively base-2).
#define ASCALE (0.125f * 1.44269504088896340736f)

typedef __attribute__((ext_vector_type(16))) _Float16 v16h;
typedef __attribute__((ext_vector_type(8)))  float    v8f;

union frag16 {
    v16h v;
    uint4 u[2];
};

// Native base-2 transcendentals (v_exp_f32 / v_log_f32).
__device__ inline float fast_exp2(float x) { return __builtin_amdgcn_exp2f(x); }
__device__ inline float fast_log2(float x) { return __builtin_amdgcn_logf(x); }

// ---------------------------------------------------------------------------
// Fragment loaders for V_WMMA_F32_16X16X32_F16 from row-major (N x 64) f16.
// A (16x32, MxK): lane l holds M = l%16;
//   halves h -> K = c0 + 16*(l/16) + h
// -> one contiguous 16-half (32B) run = two 16B loads.
// B (32x16, KxN): lane l holds N = l%16; halves h -> K = c0 + 16*(l/16) + h
// -> one contiguous 16-half (32B) run = two 16B loads.
// ---------------------------------------------------------------------------
__device__ inline v16h load_fragA(const _Float16* __restrict__ baseT, int row,
                                  int c0, int hi) {
    frag16 f;
    const _Float16* p = baseT + (size_t)row * 64 + c0 + 16 * hi;
    f.u[0] = *reinterpret_cast<const uint4*>(p);
    f.u[1] = *reinterpret_cast<const uint4*>(p + 8);
    return f.v;
}

__device__ inline v16h load_fragB(const _Float16* __restrict__ baseT, int row,
                                  int c0, int hi) {
    frag16 f;
    const _Float16* p = baseT + (size_t)row * 64 + c0 + 16 * hi;
    f.u[0] = *reinterpret_cast<const uint4*>(p);
    f.u[1] = *reinterpret_cast<const uint4*>(p + 8);
    return f.v;
}

__device__ inline v8f wmma_f16(v16h a, v16h b, v8f c) {
    return __builtin_amdgcn_wmma_f32_16x16x32_f16(false, a, false, b,
                                                  (short)0, c, false, false);
}

// Error-compensated f16 GEMM tile: e = (ah+al)x(bh+bl) ~ ah.bh + ah.bl + al.bh
// Two independent accumulator chains interleaved to cover WMMA D->C hazards.
__device__ inline v8f energy_tile(v16h ah0, v16h ah1, v16h al0, v16h al1,
                                  v16h bh0, v16h bh1, v16h bl0, v16h bl1) {
    v8f s1 = {}, s2 = {};
    s1 = wmma_f16(ah0, bh0, s1);
    s2 = wmma_f16(ah1, bh1, s2);
    s1 = wmma_f16(ah0, bl0, s1);
    s2 = wmma_f16(ah1, bl1, s2);
    s1 = wmma_f16(al0, bh0, s1);
    s2 = wmma_f16(al1, bh1, s2);
    return s1 + s2;
}

// ---------------------------------------------------------------------------
// Kernel 1: dep' = relu(bn1(conv1x1(dep, conv_w)))   (fp32)
// grid (NS/256, 2, NB), block 256. blockIdx.y selects 32 output channels.
// ---------------------------------------------------------------------------
__global__ __launch_bounds__(256)
void k_conv_bn_relu(const float* __restrict__ dep, const float* __restrict__ cw,
                    const float* __restrict__ g, const float* __restrict__ bt,
                    const float* __restrict__ mu, const float* __restrict__ vr,
                    float* __restrict__ depP) {
    const int b = blockIdx.z;
    const int obase = blockIdx.y * 32;
    const int n = blockIdx.x * 256 + threadIdx.x;

    __shared__ float wsm[32][C1];
    for (int k = threadIdx.x; k < 32 * C1; k += 256)
        wsm[k >> 8][k & 255] = cw[(obase + (k >> 8)) * C1 + (k & 255)];
    __syncthreads();

    float acc[32];
#pragma unroll
    for (int o = 0; o < 32; ++o) acc[o] = 0.f;

    const float* depb = dep + (size_t)b * C1 * NS;
    for (int c = 0; c < C1; ++c) {
        float dv = depb[(size_t)c * NS + n];
#pragma unroll
        for (int o = 0; o < 32; ++o) acc[o] = fmaf(wsm[o][c], dv, acc[o]);
    }
#pragma unroll
    for (int o = 0; o < 32; ++o) {
        int oc = obase + o;
        float s = g[oc] / sqrtf(vr[oc] + EPS);
        float bb = bt[oc] - mu[oc] * s;
        depP[((size_t)b * C2 + oc) * NS + n] = fmaxf(fmaf(acc[o], s, bb), 0.f);
    }
}

// ---------------------------------------------------------------------------
// Kernel 2: proj_rgb = rgb_w@rgb ; proj_dep = dep_w@dep'
// Writes hi/lo f16 splits, transposed (N x 64); rgb side pre-scaled by
// ASCALE = log2(e)/sqrt(C2) so the WMMA result is a base-2 logit.
// Also writes fp32 V = proj_rgb + proj_dep in (C2 x N) layout.
// grid (NS/256, 2, NB), block 256.
// ---------------------------------------------------------------------------
__global__ __launch_bounds__(256)
void k_proj(const float* __restrict__ rgb, const float* __restrict__ depP,
            const float* __restrict__ rw, const float* __restrict__ dw,
            _Float16* __restrict__ prTh, _Float16* __restrict__ prTl,
            _Float16* __restrict__ pdTh, _Float16* __restrict__ pdTl,
            float* __restrict__ V) {
    const int b = blockIdx.z;
    const int obase = blockIdx.y * 32;
    const int n = blockIdx.x * 256 + threadIdx.x;

    __shared__ float wr[32][C2];
    __shared__ float wd[32][C2];
    for (int k = threadIdx.x; k < 32 * C2; k += 256) {
        int o = k >> 6, c = k & 63;
        wr[o][c] = rw[(obase + o) * C2 + c];
        wd[o][c] = dw[(obase + o) * C2 + c];
    }
    __syncthreads();

    float ar[32], ad[32];
#pragma unroll
    for (int o = 0; o < 32; ++o) { ar[o] = 0.f; ad[o] = 0.f; }

    const float* rgbb = rgb + (size_t)b * C2 * NS;
    const float* depb = depP + (size_t)b * C2 * NS;
    for (int c = 0; c < C2; ++c) {
        float rv = rgbb[(size_t)c * NS + n];
        float dv = depb[(size_t)c * NS + n];
#pragma unroll
        for (int o = 0; o < 32; ++o) {
            ar[o] = fmaf(wr[o][c], rv, ar[o]);
            ad[o] = fmaf(wd[o][c], dv, ad[o]);
        }
    }
    const size_t rowOff = ((size_t)b * NS + n) * 64 + obase;
#pragma unroll
    for (int o = 0; o < 32; ++o) {
        float xr = ar[o] * ASCALE;               // fold scale into K operand
        _Float16 xrh = (_Float16)xr;
        prTh[rowOff + o] = xrh;
        prTl[rowOff + o] = (_Float16)(xr - (float)xrh);
        float xd = ad[o];
        _Float16 xdh = (_Float16)xd;
        pdTh[rowOff + o] = xdh;
        pdTl[rowOff + o] = (_Float16)(xd - (float)xdh);
        V[((size_t)b * C2 + obase + o) * NS + n] = ar[o] + ad[o];
    }
}

// ---------------------------------------------------------------------------
// Kernel 3 (pass 1): per-column softmax stats over i (axis=1), base-2 domain.
// Each wave owns 16 columns j; B-fragments are loop-invariant; loop over i
// tiles doing 6x v_wmma_f32_16x16x32_f16 (split-f16), online max/sumexp held
// PER LANE (no cross-lane traffic in the loop); lane^16 states merged once.
// Emits fused d_j = m_j + log2(l_j) so pass 2 needs a single stream.
// grid (NS/64, NB), block 128 (4 waves).
// ---------------------------------------------------------------------------
__global__ __launch_bounds__(128)
void k_attn_colstats(const _Float16* __restrict__ prTh,
                     const _Float16* __restrict__ prTl,
                     const _Float16* __restrict__ pdTh,
                     const _Float16* __restrict__ pdTl,
                     float* __restrict__ dOut) {
    const int b = blockIdx.y;
    const int wave = threadIdx.x >> 5;
    const int lane = threadIdx.x & 31;
    const int l16 = lane & 15, hi = lane >> 4;
    const int j0 = blockIdx.x * 64 + wave * 16;
    const size_t boff = (size_t)b * NS * 64;

    const v16h bh0 = load_fragB(pdTh + boff, j0 + l16, 0, hi);
    const v16h bh1 = load_fragB(pdTh + boff, j0 + l16, 32, hi);
    const v16h bl0 = load_fragB(pdTl + boff, j0 + l16, 0, hi);
    const v16h bl1 = load_fragB(pdTl + boff, j0 + l16, 32, hi);

    float m_cur = -INFINITY, l_cur = 0.f;
    for (int i0 = 0; i0 < NS; i0 += 16) {
        v16h ah0 = load_fragA(prTh + boff, i0 + l16, 0, hi);
        v16h ah1 = load_fragA(prTh + boff, i0 + l16, 32, hi);
        v16h al0 = load_fragA(prTl + boff, i0 + l16, 0, hi);
        v16h al1 = load_fragA(prTl + boff, i0 + l16, 32, hi);
        v8f s = energy_tile(ah0, ah1, al0, al1, bh0, bh1, bl0, bl1);

        float tmax = s[0];
#pragma unroll
        for (int p = 1; p < 8; ++p) tmax = fmaxf(tmax, s[p]);
        float m_new = fmaxf(m_cur, tmax);

        float se = 0.f;
#pragma unroll
        for (int p = 0; p < 8; ++p) se += fast_exp2(s[p] - m_new);

        l_cur = l_cur * fast_exp2(m_cur - m_new) + se;
        m_cur = m_new;
    }
    // merge the two half-wave online-softmax states for the same column
    float m_o = __shfl_xor(m_cur, 16, 32);
    float l_o = __shfl_xor(l_cur, 16, 32);
    float m_f = fmaxf(m_cur, m_o);
    float l_f = l_cur * fast_exp2(m_cur - m_f) + l_o * fast_exp2(m_o - m_f);
    if (hi == 0)
        dOut[(size_t)b * NS + j0 + l16] = m_f + fast_log2(l_f);
}

// ---------------------------------------------------------------------------
// Kernel 4 (pass 2): w[i] = (1/N) * sum_j 2^(e_ij - d_j).
// Each wave owns 16 rows i; A-fragments loop-invariant; loop over j tiles.
// grid (NS/64, NB), block 128.
// ---------------------------------------------------------------------------
__global__ __launch_bounds__(128)
void k_attn_rowweights(const _Float16* __restrict__ prTh,
                       const _Float16* __restrict__ prTl,
                       const _Float16* __restrict__ pdTh,
                       const _Float16* __restrict__ pdTl,
                       const float* __restrict__ dIn,
                       float* __restrict__ wOut) {
    const int b = blockIdx.y;
    const int wave = threadIdx.x >> 5;
    const int lane = threadIdx.x & 31;
    const int l16 = lane & 15, hi = lane >> 4;
    const int i0 = blockIdx.x * 64 + wave * 16;
    const size_t boff = (size_t)b * NS * 64;
    const float* db = dIn + (size_t)b * NS;

    const v16h ah0 = load_fragA(prTh + boff, i0 + l16, 0, hi);
    const v16h ah1 = load_fragA(prTh + boff, i0 + l16, 32, hi);
    const v16h al0 = load_fragA(prTl + boff, i0 + l16, 0, hi);
    const v16h al1 = load_fragA(prTl + boff, i0 + l16, 32, hi);

    float acc8[8];
#pragma unroll
    for (int p = 0; p < 8; ++p) acc8[p] = 0.f;

    for (int j0 = 0; j0 < NS; j0 += 16) {
        v16h bh0 = load_fragB(pdTh + boff, j0 + l16, 0, hi);
        v16h bh1 = load_fragB(pdTh + boff, j0 + l16, 32, hi);
        v16h bl0 = load_fragB(pdTl + boff, j0 + l16, 0, hi);
        v16h bl1 = load_fragB(pdTl + boff, j0 + l16, 32, hi);
        v8f s = energy_tile(ah0, ah1, al0, al1, bh0, bh1, bl0, bl1);

        float dj = db[j0 + l16];
#pragma unroll
        for (int p = 0; p < 8; ++p) acc8[p] += fast_exp2(s[p] - dj);
    }
    // sum over the 16 columns held across lanes of each half-wave
#pragma unroll
    for (int off = 1; off < 16; off <<= 1) {
#pragma unroll
        for (int p = 0; p < 8; ++p) acc8[p] += __shfl_xor(acc8[p], off, 32);
    }
    if (l16 == 0) {
#pragma unroll
        for (int p = 0; p < 8; ++p)
            wOut[(size_t)b * NS + i0 + p + 8 * hi] = acc8[p] * (1.0f / NS);
    }
}

// ---------------------------------------------------------------------------
// Kernel 5: gap[b,c] = mean_n(rgb+dep') + sum_i V[c,i] * w[i]
// grid (C2, NB), block 256.
// ---------------------------------------------------------------------------
__global__ __launch_bounds__(256)
void k_gap(const float* __restrict__ rgb, const float* __restrict__ depP,
           const float* __restrict__ V, const float* __restrict__ w,
           float* __restrict__ gap) {
    const int b = blockIdx.y, c = blockIdx.x;
    const float* r = rgb + ((size_t)b * C2 + c) * NS;
    const float* d = depP + ((size_t)b * C2 + c) * NS;
    const float* v = V + ((size_t)b * C2 + c) * NS;
    const float* wb = w + (size_t)b * NS;

    float s1 = 0.f, s2 = 0.f;
    for (int n = threadIdx.x; n < NS; n += 256) {
        s1 += r[n] + d[n];
        s2 = fmaf(v[n], wb[n], s2);
    }
    float t = s1 * (1.0f / NS) + s2;
#pragma unroll
    for (int off = 16; off > 0; off >>= 1) t += __shfl_xor(t, off, 32);

    __shared__ float red[8];
    if ((threadIdx.x & 31) == 0) red[threadIdx.x >> 5] = t;
    __syncthreads();
    if (threadIdx.x == 0) {
        float s = 0.f;
#pragma unroll
        for (int i = 0; i < 8; ++i) s += red[i];
        gap[b * C2 + c] = s;
    }
}

// ---------------------------------------------------------------------------
// Kernel 6: gate = sigmoid(bn3(mlp2 @ relu(bn2(mlp1 @ gap))))
// grid NB, block 64.
// ---------------------------------------------------------------------------
__global__ __launch_bounds__(64)
void k_mlp(const float* __restrict__ gap, const float* __restrict__ m1,
           const float* __restrict__ g2, const float* __restrict__ b2,
           const float* __restrict__ mu2, const float* __restrict__ v2,
           const float* __restrict__ m2, const float* __restrict__ g3,
           const float* __restrict__ b3, const float* __restrict__ mu3,
           const float* __restrict__ v3, float* __restrict__ out) {
    const int b = blockIdx.x, t = threadIdx.x;
    __shared__ float gp[C2];
    __shared__ float h[HID];
    gp[t] = gap[b * C2 + t];
    __syncthreads();
    if (t < HID) {
        float a = 0.f;
        for (int c = 0; c < C2; ++c) a = fmaf(m1[t * C2 + c], gp[c], a);
        float s = g2[t] / sqrtf(v2[t] + EPS);
        h[t] = fmaxf((a - mu2[t]) * s + b2[t], 0.f);
    }
    __syncthreads();
    float a = 0.f;
    for (int k = 0; k < HID; ++k) a = fmaf(m2[t * HID + k], h[k], a);
    float s = g3[t] / sqrtf(v3[t] + EPS);
    float val = (a - mu3[t]) * s + b3[t];
    out[b * C2 + t] = 1.0f / (1.0f + __expf(-val));
}

// ---------------------------------------------------------------------------
extern "C" void kernel_launch(void* const* d_in, const int* in_sizes, int n_in,
                              void* d_out, int out_size, void* d_ws,
                              size_t ws_size, hipStream_t stream) {
    const float* rgb    = (const float*)d_in[0];
    const float* dep    = (const float*)d_in[1];
    const float* conv_w = (const float*)d_in[2];
    const float* bn1_g  = (const float*)d_in[3];
    const float* bn1_b  = (const float*)d_in[4];
    const float* bn1_m  = (const float*)d_in[5];
    const float* bn1_v  = (const float*)d_in[6];
    const float* rgb_w  = (const float*)d_in[7];
    const float* dep_w  = (const float*)d_in[8];
    const float* mlp1_w = (const float*)d_in[9];
    const float* bn2_g  = (const float*)d_in[10];
    const float* bn2_b  = (const float*)d_in[11];
    const float* bn2_m  = (const float*)d_in[12];
    const float* bn2_v  = (const float*)d_in[13];
    const float* mlp2_w = (const float*)d_in[14];
    const float* bn3_g  = (const float*)d_in[15];
    const float* bn3_b  = (const float*)d_in[16];
    const float* bn3_m  = (const float*)d_in[17];
    const float* bn3_v  = (const float*)d_in[18];
    float* out = (float*)d_out;

    char* ws = (char*)d_ws;
    float*    depP = (float*)(ws);                     // 4 MB
    float*    V    = (float*)(ws + (4u << 20));        // 4 MB
    _Float16* prTh = (_Float16*)(ws + (8u << 20));     // 2 MB (scaled)
    _Float16* prTl = (_Float16*)(ws + (10u << 20));    // 2 MB
    _Float16* pdTh = (_Float16*)(ws + (12u << 20));    // 2 MB
    _Float16* pdTl = (_Float16*)(ws + (14u << 20));    // 2 MB
    float*    dA   = (float*)(ws + (16u << 20));       // 64 KB (m + log2 l)
    float*    wA   = dA + NB * NS;                     // 64 KB
    float*    gapA = wA + NB * NS;                     // 1 KB

    k_conv_bn_relu<<<dim3(NS / 256, 2, NB), 256, 0, stream>>>(
        dep, conv_w, bn1_g, bn1_b, bn1_m, bn1_v, depP);

    k_proj<<<dim3(NS / 256, 2, NB), 256, 0, stream>>>(
        rgb, depP, rgb_w, dep_w, prTh, prTl, pdTh, pdTl, V);

    k_attn_colstats<<<dim3(NS / 64, NB), 128, 0, stream>>>(
        prTh, prTl, pdTh, pdTl, dA);

    k_attn_rowweights<<<dim3(NS / 64, NB), 128, 0, stream>>>(
        prTh, prTl, pdTh, pdTl, dA, wA);

    k_gap<<<dim3(C2, NB), 256, 0, stream>>>(rgb, depP, V, wA, gapA);

    k_mlp<<<dim3(NB), 64, 0, stream>>>(gapA, mlp1_w, bn2_g, bn2_b, bn2_m,
                                       bn2_v, mlp2_w, bn3_g, bn3_b, bn3_m,
                                       bn3_v, out);
}
